// MambaModel_72730976190812
// MI455X (gfx1250) — matrix operand order repaired and checked by reading, not compiled, not run on hardware
//
#include <hip/hip_runtime.h>
#include <hip/hip_bf16.h>
#include <math.h>

// ---- model dims (match reference) ----
#define VOCAB   32000
#define D_MODEL 768
#define N_LAYERS 4
#define STATE   16
#define D_CONV  4
#define D_INNER 1536      // 2 * 768
#define DT_RANK 96        // ceil(1536/16)
#define DT_PAD  128       // DT_RANK zero-padded to a multiple of 64 for the GEMM
#define SEQ     2048
#define D_XPROJ (DT_RANK + 2*STATE)   // 128

#define SCAN_CH 16
#define CHLEN   (SEQ / SCAN_CH)       // 128
#define NCHAN   (D_INNER * STATE)     // 24576

typedef __bf16 bf16;
typedef __attribute__((ext_vector_type(16))) __bf16 v16bf;
typedef __attribute__((ext_vector_type(8)))  __bf16 v8bf;
typedef __attribute__((ext_vector_type(8)))  float  v8f;
typedef int v4i __attribute__((vector_size(16)));
typedef unsigned int u32x4 __attribute__((vector_size(16)));
typedef int i32x4 __attribute__((vector_size(16)));
typedef int i32x8 __attribute__((vector_size(32)));

#if __has_builtin(__builtin_amdgcn_global_load_async_to_lds_b128)
#define HAVE_ASYNC_LDS 1
#endif
#if __has_builtin(__builtin_amdgcn_tensor_load_to_lds) && __has_builtin(__builtin_amdgcn_s_wait_tensorcnt)
#define HAVE_TDM 1
#endif

template <int N>
__device__ __forceinline__ void wait_async() {
#if __has_builtin(__builtin_amdgcn_s_wait_asynccnt)
    __builtin_amdgcn_s_wait_asynccnt(N);
#else
    asm volatile("s_wait_asynccnt %0" :: "i"(N) : "memory");
#endif
}

template <int N>
__device__ __forceinline__ void wait_tensor() {
#if __has_builtin(__builtin_amdgcn_s_wait_tensorcnt)
    __builtin_amdgcn_s_wait_tensorcnt(N);
#else
    asm volatile("s_wait_tensorcnt %0" :: "i"(N) : "memory");
#endif
}

#ifdef HAVE_TDM
// Issue one TDM 2D tile load: rows x 64-col bf16 tile -> LDS (+hardware row pad).
// D# group0: count=1 | lds_addr | global_addr | type=2.
// D# group1: data_size=2B, pad_enable (32 DW interval, 4 DW amount -> row 128B+16B),
//            tensor_dim0=dim0, tensor_dim1=rows, tile 64x64, dim0_stride=stride.
__device__ __forceinline__ void tdm_issue(unsigned lds_off, const bf16* gp,
                                          int dim0, int rows, int stride) {
    unsigned long long ga = (unsigned long long)(size_t)gp;
    u32x4 g0 = { 1u, lds_off, (unsigned)ga,
                 (unsigned)((ga >> 32) & 0x01ffffffu) | 0x80000000u };
    i32x8 g1 = {
        (int)((1u << 16) | (1u << 20) | (4u << 22) | (3u << 25)),
        (int)((dim0 & 0xffff) << 16),
        (int)(((dim0 >> 16) & 0xffff) | ((rows & 0xffff) << 16)),
        (int)(((rows >> 16) & 0xffff) | (64 << 16)),   // tile_dim0 = 64
        64,                                            // tile_dim1 = 64, tile_dim2 = 0
        stride,                                        // tensor_dim0_stride[31:0]
        0, 0
    };
    i32x4 gz = { 0, 0, 0, 0 };
#if __clang_major__ >= 23
    i32x8 gz8 = { 0, 0, 0, 0, 0, 0, 0, 0 };
    __builtin_amdgcn_tensor_load_to_lds(g0, g1, gz, gz, gz8, 0);
#else
    __builtin_amdgcn_tensor_load_to_lds(g0, g1, gz, gz, 0);
#endif
}
#endif

// ======================= elementwise / prep kernels =======================

__global__ void k_cast_bf16(const float* __restrict__ src, bf16* __restrict__ dst, long n) {
    long i = (long)blockIdx.x * blockDim.x + threadIdx.x;
    if (i < n) dst[i] = (bf16)src[i];
}

// cast rows of length ksrc into zero-padded rows of length kdst
__global__ void k_cast_pad_bf16(const float* __restrict__ src, bf16* __restrict__ dst,
                                long rows, int ksrc, int kdst) {
    long i = (long)blockIdx.x * blockDim.x + threadIdx.x;
    if (i >= rows * kdst) return;
    int k = (int)(i % kdst);
    long r = i / kdst;
    dst[i] = (k < ksrc) ? (bf16)src[r * ksrc + k] : (bf16)0.f;
}

__global__ void k_embed(const int* __restrict__ ids, const float* __restrict__ emb,
                        float* __restrict__ x) {
    int l = blockIdx.x;
    long id = ids[l];
    const float* src = emb + id * (long)D_MODEL;
    float* dst = x + (long)l * D_MODEL;
    for (int d = threadIdx.x; d < D_MODEL; d += blockDim.x) dst[d] = src[d];
}

__global__ void __launch_bounds__(256)
k_rmsnorm_bf16(const float* __restrict__ x, const float* __restrict__ w,
               bf16* __restrict__ out) {
    int l = blockIdx.x;
    const float* row = x + (long)l * D_MODEL;
    float ss = 0.f;
    for (int d = threadIdx.x; d < D_MODEL; d += 256) { float v = row[d]; ss += v * v; }
    for (int off = 16; off; off >>= 1) ss += __shfl_xor(ss, off, 32);
    __shared__ float sred[8];
    int wave = threadIdx.x >> 5, lane = threadIdx.x & 31;
    if (lane == 0) sred[wave] = ss;
    __syncthreads();
    if (wave == 0) {
        float t = (lane < 8) ? sred[lane] : 0.f;
        for (int off = 4; off; off >>= 1) t += __shfl_xor(t, off, 32);
        if (lane == 0) sred[0] = t;
    }
    __syncthreads();
    float inv = rsqrtf(sred[0] / (float)D_MODEL + 1e-5f);
    bf16* o = out + (long)l * D_MODEL;
    for (int d = threadIdx.x; d < D_MODEL; d += 256)
        o[d] = (bf16)(row[d] * inv * w[d]);
}

__global__ void k_conv_silu(const float* __restrict__ xz, const float* __restrict__ w,
                            const float* __restrict__ b, float* __restrict__ xc,
                            bf16* __restrict__ xcb) {
    long i = (long)blockIdx.x * blockDim.x + threadIdx.x;
    if (i >= (long)SEQ * D_INNER) return;
    int c = (int)(i % D_INNER);
    int l = (int)(i / D_INNER);
    float acc = b[c];
#pragma unroll
    for (int j = 0; j < D_CONV; ++j) {
        int t = l - (D_CONV - 1) + j;
        if (t >= 0) acc += w[c * D_CONV + j] * xz[(long)t * (2 * D_INNER) + c];
    }
    float s = acc / (1.f + expf(-acc));
    xc[i] = s;
    xcb[i] = (bf16)s;
}

// pack dt_inter slice of proj[SEQ][128] into zero-padded bf16 [SEQ][DT_PAD]
__global__ void k_dtpack(const float* __restrict__ proj, bf16* __restrict__ dtb) {
    long i = (long)blockIdx.x * blockDim.x + threadIdx.x;
    if (i >= (long)SEQ * DT_PAD) return;
    int r = (int)(i % DT_PAD);
    int l = (int)(i / DT_PAD);
    dtb[i] = (r < DT_RANK) ? (bf16)proj[(long)l * D_XPROJ + r] : (bf16)0.f;
}

__global__ void k_softplus(float* __restrict__ dtg, const float* __restrict__ dtb) {
    long i = (long)blockIdx.x * blockDim.x + threadIdx.x;
    if (i >= (long)SEQ * D_INNER) return;
    float v = dtg[i] + dtb[(int)(i % D_INNER)];
    dtg[i] = (v > 20.f) ? v : log1pf(expf(v));
}

// ======================= chunked selective scan =======================
__global__ void __launch_bounds__(256)
k_scan_part1(const float* __restrict__ delta, const float* __restrict__ proj,
             const float* __restrict__ xc, const float* __restrict__ A_log,
             float* __restrict__ P, float* __restrict__ Ho) {
    int r = blockIdx.x * 256 + threadIdx.x;
    int c = blockIdx.y;
    int d = r >> 4, n = r & 15;
    float A = -expf(A_log[r]);
    float h = 0.f, Pr = 1.f;
    int l0 = c * CHLEN;
    for (int l = l0; l < l0 + CHLEN; ++l) {
        float dl = delta[(long)l * D_INNER + d];
        float Bn = proj[(long)l * D_XPROJ + DT_RANK + n];
        float xv = xc[(long)l * D_INNER + d];
        float dA = expf(fminf(fmaxf(dl * A, -7.f), 20.f));
        h = fmaf(dA, h, dl * Bn * xv);
        Pr *= dA;
    }
    P[(long)c * NCHAN + r]  = Pr;
    Ho[(long)c * NCHAN + r] = h;
}

__global__ void k_scan_part2(const float* __restrict__ P, const float* __restrict__ Ho,
                             float* __restrict__ Hin) {
    int r = blockIdx.x * 256 + threadIdx.x;
    if (r >= NCHAN) return;
    float H = 0.f;
#pragma unroll
    for (int c = 0; c < SCAN_CH; ++c) {
        Hin[(long)c * NCHAN + r] = H;
        H = fmaf(P[(long)c * NCHAN + r], H, Ho[(long)c * NCHAN + r]);
    }
}

__global__ void __launch_bounds__(256)
k_scan_part3(const float* __restrict__ delta, const float* __restrict__ proj,
             const float* __restrict__ xc, const float* __restrict__ xz,
             const float* __restrict__ A_log, const float* __restrict__ Dp,
             const float* __restrict__ Hin, bf16* __restrict__ yb) {
    int r = blockIdx.x * 256 + threadIdx.x;
    int c = blockIdx.y;
    int d = r >> 4, n = r & 15;
    float A = -expf(A_log[r]);
    float Dd = Dp[d];
    float h = Hin[(long)c * NCHAN + r];
    int l0 = c * CHLEN;
    for (int l = l0; l < l0 + CHLEN; ++l) {
        float dl = delta[(long)l * D_INNER + d];
        float Bn = proj[(long)l * D_XPROJ + DT_RANK + n];
        float Cn = proj[(long)l * D_XPROJ + DT_RANK + STATE + n];
        float xv = xc[(long)l * D_INNER + d];
        float dA = expf(fminf(fmaxf(dl * A, -7.f), 20.f));
        h = fmaf(dA, h, dl * Bn * xv);
        float p = Cn * h;
        p += __shfl_xor(p, 1, 16);
        p += __shfl_xor(p, 2, 16);
        p += __shfl_xor(p, 4, 16);
        p += __shfl_xor(p, 8, 16);
        if (n == 0) {
            float yv = p + xv * Dd;
            float z = xz[(long)l * (2 * D_INNER) + D_INNER + d];
            float s = z / (1.f + expf(-z));
            yb[(long)l * D_INNER + d] = (bf16)(yv * s);
        }
    }
}

// ======================= WMMA bf16 GEMM =======================
// C[M][N](f32) = A[M][K](bf16 rm) @ W[N][K](bf16 rm)^T  (+ optional addsrc)
// block 256 thr (8 waves); block tile 256x64; wave tile 32x64.
// 64-wide K-slab of W staged in LDS by TDM (wave 0, one descriptor per slab),
// double buffered; falls back to per-lane async-to-LDS loads.

#define KS   64   // K-slab
#define LROW 72   // padded LDS row stride (128B data + 16B pad = 36 banks)

__device__ __forceinline__ v16bf frag_a(const bf16* __restrict__ base, int ld, int k0) {
    int lane = threadIdx.x & 31;
    int half = lane >> 4;
    const bf16* p = base + (size_t)(lane & 15) * ld + k0;
    v8bf lo = *(const v8bf*)(p + half * 8);
    v8bf hi = *(const v8bf*)(p + 16 + half * 8);
    v16bf f;
#pragma unroll
    for (int j = 0; j < 8; ++j) { f[j] = lo[j]; f[8 + j] = hi[j]; }
    return f;
}

__global__ void __launch_bounds__(256)
k_gemm_bf16(const bf16* __restrict__ A, const bf16* __restrict__ W,
            const float* __restrict__ addsrc, float* __restrict__ C,
            int M, int N, int K) {
    __shared__ __align__(16) bf16 shW[2][64 * LROW];

    int wave = threadIdx.x >> 5;
    int m0 = blockIdx.y * 256 + wave * 32;
    int n0 = blockIdx.x * 64;
    const bf16* A0 = A + (size_t)m0 * K;
    const bf16* A1 = A + (size_t)(m0 + 16) * K;

    auto stage = [&](int kk, int buf) {
#ifdef HAVE_TDM
        if (wave == 0) {
            unsigned lds_off =
                (unsigned)(size_t)(__attribute__((address_space(3))) void*)&shW[buf][0];
            tdm_issue(lds_off, W + (size_t)n0 * K + kk, K - kk, N - n0, K);
        }
#elif defined(HAVE_ASYNC_LDS)
        int t = threadIdx.x;
        int row = t >> 2, seg = t & 3;               // 4 threads x 32B per 128B row
        const bf16* g = W + (size_t)(n0 + row) * K + kk + seg * 16;
        bf16* l = &shW[buf][row * LROW + seg * 16];
        __builtin_amdgcn_global_load_async_to_lds_b128(
            (__attribute__((address_space(1))) v4i*)(bf16*)g,
            (__attribute__((address_space(3))) v4i*)(bf16*)l, 0, 0);
        __builtin_amdgcn_global_load_async_to_lds_b128(
            (__attribute__((address_space(1))) v4i*)(bf16*)(g + 8),
            (__attribute__((address_space(3))) v4i*)(bf16*)(l + 8), 0, 0);
#else
        int t = threadIdx.x;
        int row = t >> 2, seg = t & 3;
        const bf16* g = W + (size_t)(n0 + row) * K + kk + seg * 16;
        bf16* l = &shW[buf][row * LROW + seg * 16];
        *(v8bf*)l = *(const v8bf*)g;
        *(v8bf*)(l + 8) = *(const v8bf*)(g + 8);
#endif
    };
    auto wait_stage = [&](bool more) {
#ifdef HAVE_TDM
        if (wave == 0) { if (more) wait_tensor<1>(); else wait_tensor<0>(); }
#elif defined(HAVE_ASYNC_LDS)
        if (more) wait_async<2>(); else wait_async<0>();
#else
        asm volatile("s_wait_dscnt 0x0" ::: "memory");
#endif
    };
    // B fragment (32x16 KxN) from LDS slab: sub-step ss selects k 0..31 / 32..63
    auto frag_b_lds = [&](int buf, int ss, int c) -> v16bf {
        int lane = threadIdx.x & 31;
        int half = lane >> 4;
        const bf16* p = &shW[buf][(c * 16 + (lane & 15)) * LROW + ss * 32];
        v8bf lo = *(const v8bf*)(p + half * 8);
        v8bf hi = *(const v8bf*)(p + 16 + half * 8);
        v16bf f;
#pragma unroll
        for (int j = 0; j < 8; ++j) { f[j] = lo[j]; f[8 + j] = hi[j]; }
        return f;
    };

    int nslabs = K >> 6;                 // K is a multiple of 64 for all call sites
    stage(0, 0);
    v16bf a00 = frag_a(A0, K, 0),  a01 = frag_a(A1, K, 0);
    v16bf a10 = frag_a(A0, K, 32), a11 = frag_a(A1, K, 32);
    v8f acc[8] = {};

    int lane15 = threadIdx.x & 15;
    for (int s = 0; s < nslabs; ++s) {
        int cur = s & 1;
        bool more = (s + 1 < nslabs);
        if (more) {
            stage((s + 1) << 6, cur ^ 1);
            __builtin_prefetch(A0 + (size_t)lane15 * K + ((s + 1) << 6), 0, 1);
        }
        wait_stage(more);
        __builtin_amdgcn_s_barrier();

        v16bf b00 = a00, b01 = a01, b10 = a10, b11 = a11;
        if (more) {
            int kn = (s + 1) << 6;
            b00 = frag_a(A0, K, kn);      b01 = frag_a(A1, K, kn);
            b10 = frag_a(A0, K, kn + 32); b11 = frag_a(A1, K, kn + 32);
        }
#pragma unroll
        for (int c = 0; c < 4; ++c) {
            v16bf b = frag_b_lds(cur, 0, c);
            acc[c]     = __builtin_amdgcn_wmma_f32_16x16x32_bf16(false, a00, false, b, (short)0, acc[c],     false, false);
            acc[4 + c] = __builtin_amdgcn_wmma_f32_16x16x32_bf16(false, a01, false, b, (short)0, acc[4 + c], false, false);
        }
#pragma unroll
        for (int c = 0; c < 4; ++c) {
            v16bf b = frag_b_lds(cur, 1, c);
            acc[c]     = __builtin_amdgcn_wmma_f32_16x16x32_bf16(false, a10, false, b, (short)0, acc[c],     false, false);
            acc[4 + c] = __builtin_amdgcn_wmma_f32_16x16x32_bf16(false, a11, false, b, (short)0, acc[4 + c], false, false);
        }
        __builtin_amdgcn_s_barrier();
        a00 = b00; a01 = b01; a10 = b10; a11 = b11;
    }

    int lane = threadIdx.x & 31;
    int half = lane >> 4;
    int nn = lane & 15;
#pragma unroll
    for (int rstrip = 0; rstrip < 2; ++rstrip) {
#pragma unroll
        for (int c = 0; c < 4; ++c) {
#pragma unroll
            for (int e = 0; e < 8; ++e) {
                size_t idx = (size_t)(m0 + rstrip * 16 + half * 8 + e) * N + (n0 + c * 16 + nn);
                float v = acc[rstrip * 4 + c][e];
                if (addsrc) v += addsrc[idx];
                C[idx] = v;
            }
        }
    }
}

// ======================= host-side launcher =======================
static inline long cdivl(long a, long b) { return (a + b - 1) / b; }

extern "C" void kernel_launch(void* const* d_in, const int* in_sizes, int n_in,
                              void* d_out, int out_size, void* d_ws, size_t ws_size,
                              hipStream_t stream) {
    (void)in_sizes; (void)n_in; (void)out_size; (void)ws_size;
    const int*   ids   = (const int*)  d_in[0];
    const float* embW  = (const float*)d_in[1];
    const float* normw = (const float*)d_in[2];
    const float* inW   = (const float*)d_in[3];
    const float* convW = (const float*)d_in[4];
    const float* convb = (const float*)d_in[5];
    const float* xpW   = (const float*)d_in[6];
    const float* dtW   = (const float*)d_in[7];
    const float* dtb   = (const float*)d_in[8];
    const float* Alog  = (const float*)d_in[9];
    const float* Dp    = (const float*)d_in[10];
    const float* outW  = (const float*)d_in[11];
    const float* normf = (const float*)d_in[12];
    float* logits = (float*)d_out;

    char* wsp = (char*)d_ws;
    auto alloc = [&](size_t bytes) -> char* {
        char* p = wsp; wsp += (bytes + 255) & ~(size_t)255; return p;
    };
    const long N_EMB = (long)VOCAB * D_MODEL;
    const long N_INP = (long)2 * D_INNER * D_MODEL;
    const long N_XPR = (long)D_XPROJ * D_INNER;
    const long N_DTW = (long)D_INNER * DT_PAD;        // padded layout
    const long N_OUT = (long)D_MODEL * D_INNER;

    bf16* embB  = (bf16*)alloc(N_EMB * 2);
    bf16* inWB  = (bf16*)alloc(N_LAYERS * N_INP * 2);
    bf16* xpWB  = (bf16*)alloc(N_LAYERS * N_XPR * 2);
    bf16* dtWB  = (bf16*)alloc(N_LAYERS * N_DTW * 2);
    bf16* outWB = (bf16*)alloc(N_LAYERS * N_OUT * 2);

    float* x     = (float*)alloc((long)SEQ * D_MODEL * 4);
    bf16*  xnB   = (bf16*) alloc((long)SEQ * D_MODEL * 2);
    float* xz    = (float*)alloc((long)SEQ * 2 * D_INNER * 4);
    float* xcf   = (float*)alloc((long)SEQ * D_INNER * 4);
    bf16*  xcB   = (bf16*) alloc((long)SEQ * D_INNER * 2);
    float* proj  = (float*)alloc((long)SEQ * D_XPROJ * 4);
    bf16*  dtinB = (bf16*) alloc((long)SEQ * DT_PAD * 2);
    float* dtg   = (float*)alloc((long)SEQ * D_INNER * 4);
    bf16*  ybB   = (bf16*) alloc((long)SEQ * D_INNER * 2);
    float* Pbuf  = (float*)alloc((long)SCAN_CH * NCHAN * 4);
    float* Hobuf = (float*)alloc((long)SCAN_CH * NCHAN * 4);
    float* Hin   = (float*)alloc((long)SCAN_CH * NCHAN * 4);

    auto cast = [&](const float* s, bf16* d, long n) {
        k_cast_bf16<<<dim3((unsigned)cdivl(n, 256)), dim3(256), 0, stream>>>(s, d, n);
    };
    cast(embW, embB, N_EMB);
    cast(inW,  inWB,  N_LAYERS * N_INP);
    cast(xpW,  xpWB,  N_LAYERS * N_XPR);
    cast(outW, outWB, N_LAYERS * N_OUT);
    // dt_W: rows of 96 -> zero-padded rows of 128
    k_cast_pad_bf16<<<dim3((unsigned)cdivl((long)N_LAYERS * D_INNER * DT_PAD, 256)),
                      dim3(256), 0, stream>>>(dtW, dtWB, (long)N_LAYERS * D_INNER,
                                              DT_RANK, DT_PAD);

    k_embed<<<dim3(SEQ), dim3(256), 0, stream>>>(ids, embW, x);

    const long nDI = (long)SEQ * D_INNER;
    dim3 blk(256);
    dim3 scan_grid(NCHAN / 256, SCAN_CH);
    for (int L = 0; L < N_LAYERS; ++L) {
        k_rmsnorm_bf16<<<dim3(SEQ), blk, 0, stream>>>(x, normw + (long)L * D_MODEL, xnB);
        k_gemm_bf16<<<dim3(2 * D_INNER / 64, SEQ / 256), blk, 0, stream>>>(
            xnB, inWB + (long)L * N_INP, nullptr, xz, SEQ, 2 * D_INNER, D_MODEL);
        k_conv_silu<<<dim3((unsigned)cdivl(nDI, 256)), blk, 0, stream>>>(
            xz, convW + (long)L * D_INNER * D_CONV, convb + (long)L * D_INNER, xcf, xcB);
        k_gemm_bf16<<<dim3(D_XPROJ / 64, SEQ / 256), blk, 0, stream>>>(
            xcB, xpWB + (long)L * N_XPR, nullptr, proj, SEQ, D_XPROJ, D_INNER);
        k_dtpack<<<dim3((unsigned)cdivl((long)SEQ * DT_PAD, 256)), blk, 0, stream>>>(proj, dtinB);
        k_gemm_bf16<<<dim3(D_INNER / 64, SEQ / 256), blk, 0, stream>>>(
            dtinB, dtWB + (long)L * N_DTW, nullptr, dtg, SEQ, D_INNER, DT_PAD);
        k_softplus<<<dim3((unsigned)cdivl(nDI, 256)), blk, 0, stream>>>(dtg, dtb + (long)L * D_INNER);
        k_scan_part1<<<scan_grid, blk, 0, stream>>>(
            dtg, proj, xcf, Alog + (long)L * NCHAN, Pbuf, Hobuf);
        k_scan_part2<<<dim3(NCHAN / 256), blk, 0, stream>>>(Pbuf, Hobuf, Hin);
        k_scan_part3<<<scan_grid, blk, 0, stream>>>(
            dtg, proj, xcf, xz, Alog + (long)L * NCHAN, Dp + (long)L * D_INNER, Hin, ybB);
        k_gemm_bf16<<<dim3(D_MODEL / 64, SEQ / 256), blk, 0, stream>>>(
            ybB, outWB + (long)L * N_OUT, x, x, SEQ, D_MODEL, D_INNER);
    }

    k_rmsnorm_bf16<<<dim3(SEQ), blk, 0, stream>>>(x, normf, xnB);
    k_gemm_bf16<<<dim3(VOCAB / 64, SEQ / 256), blk, 0, stream>>>(
        xnB, embB, nullptr, logits, SEQ, VOCAB, D_MODEL);
}
